// SchNet_56667798504233
// MI455X (gfx1250) — matrix operand order repaired and checked by reading, not compiled, not run on hardware
//
#include <hip/hip_runtime.h>
#include <hip/hip_bf16.h>

// ---------------------------------------------------------------------------
// CDNA5 (gfx1250) SchNet: fused edge cfconv + WMMA bf16 64x64 GEMMs.
// wave32: each wave owns a 16-row tile; F=64 -> K=64 (2x wmma 16x16x32),
// 4 N-tiles of 16 -> 8 v_wmma_f32_16x16x32_bf16 per 64x64 GEMM per tile.
// ---------------------------------------------------------------------------

typedef __attribute__((ext_vector_type(16))) __bf16 v16bf;
typedef __attribute__((ext_vector_type(8)))  float  v8f;

#define HS 72          // padded LDS row stride (halves) for 16x64 tiles
#define WPB 8          // waves per block (256 threads)

union Frag { v16bf v; unsigned short s[16]; };

__device__ __forceinline__ unsigned short f2bf(float f) {
  union { float f; unsigned u; } v; v.f = f;
  unsigned r = v.u + 0x7FFFu + ((v.u >> 16) & 1u);   // RNE
  return (unsigned short)(r >> 16);
}

// Branchless, stable softplus via hardware transcendentals:
// sp(x) = max(x,0) + log(1 + exp(-|x|))  -> v_exp_f32 + v_log_f32 only.
__device__ __forceinline__ float softplus_f(float x) {
  return fmaxf(x, 0.0f) + __logf(1.0f + __expf(-fabsf(x)));
}

// Intra-wave LDS stage fence: LDS ops from one wave execute in order, we only
// need to stop the compiler from reordering across the stage boundary.
__device__ __forceinline__ void wave_fence() { __builtin_amdgcn_wave_barrier(); }

// A-matrix (16x32 bf16, MxK) fragment from an LDS 16xHS tile, K-offset k0.
// ISA 7.12.2: lane lo holds row M=lo; low lanes K in {0..7,16..23}+..,
// high lanes shifted by 8. Two contiguous 8-half runs per lane.
__device__ __forceinline__ v16bf load_a(const unsigned short* hb, int lo, int hi, int k0) {
  Frag f;
  const unsigned short* row = hb + lo * HS;
#pragma unroll
  for (int e = 0; e < 8; ++e) f.s[e]     = row[k0 + hi * 8 + e];
#pragma unroll
  for (int e = 0; e < 8; ++e) f.s[8 + e] = row[k0 + 16 + hi * 8 + e];
  return f.v;
}

// B-matrix (32x16 bf16, KxN) fragment from LDS transposed weights wt[N][K]
// (64x64). Lane lo holds column N = 16*t + lo; K in {0..7,16..23} + hi*8 (matches A).
__device__ __forceinline__ v16bf load_b(const unsigned short* wt, int lo, int hi, int t, int k0) {
  Frag f;
  const unsigned short* row = wt + (t * 16 + lo) * 64;
#pragma unroll
  for (int e = 0; e < 8; ++e) f.s[e] = row[k0 + hi * 8 + e];
#pragma unroll
  for (int e = 0; e < 8; ++e) f.s[8 + e] = row[k0 + 16 + hi * 8 + e];
  return f.v;
}

__device__ __forceinline__ v8f wmma_bf16(v16bf a, v16bf b, v8f c) {
  // 8 args: (neg_a, A, neg_b, B, c_mod, C, reuse_a, reuse_b)
  return __builtin_amdgcn_wmma_f32_16x16x32_bf16(false, a, false, b, (short)0, c, false, false);
}

// ---------------------------------------------------------------------------
// Prep: convert all 19 64x64 weight matrices to bf16, transposed (wt[n][k]).
// ---------------------------------------------------------------------------
struct WSrc { const float* p[19]; };

__global__ __launch_bounds__(256) void prep_weights(WSrc ws, unsigned short* wt) {
  const float* s = ws.p[blockIdx.x];
  unsigned short* dsto = wt + (size_t)blockIdx.x * 4096;
  for (int i = threadIdx.x; i < 4096; i += 256) {
    int k = i >> 6, n = i & 63;
    dsto[n * 64 + k] = f2bf(s[i]);        // transpose: wt[n][k] = W[k][n]
  }
}

// ---------------------------------------------------------------------------
// Fused edge pass: RBF(d)*cut -> sp(.@W1+b1) -> .@W2+b2 -> *cut -> msg ->
// atomic scatter into agg[dst]. y==nullptr => embedding mode (msg = eh*filt).
// ---------------------------------------------------------------------------
__global__ __launch_bounds__(256) void edge_cfconv(
    const float* __restrict__ d, const int* __restrict__ src, const int* __restrict__ dst,
    const int* __restrict__ z, const float* __restrict__ Ecolor,
    const unsigned short* __restrict__ W1t, const float* __restrict__ b1,
    const unsigned short* __restrict__ W2t, const float* __restrict__ b2,
    const float* __restrict__ y, float* __restrict__ agg, int E) {
  __shared__ unsigned short w1[4096];
  __shared__ unsigned short w2[4096];
  __shared__ unsigned short hb[WPB][16 * HS];
  __shared__ float cutb[WPB][16];
  __shared__ int   sb[WPB][16], db[WPB][16], zsb[WPB][16], zdb[WPB][16];

  for (int i = threadIdx.x; i < 4096; i += 256) { w1[i] = W1t[i]; w2[i] = W2t[i]; }
  __syncthreads();

  const int wave = threadIdx.x >> 5, lane = threadIdx.x & 31;
  const int lo = lane & 15, hi = lane >> 4;
  const int e0 = (blockIdx.x * WPB + wave) * 16;

  // ---- per-edge scalars + radial basis (row = lo, K half = hi) ----
  const int e = e0 + lo;
  const bool ok = (e < E);
  const float de = ok ? d[e] : 2.0f;
  const float cutv =
      (de < 1.0f) ? 0.5f * (__cosf(3.14159265358979323846f * de) + 1.0f) : 0.0f;
  if (hi == 0) {
    cutb[wave][lo] = cutv;
    int sv = ok ? src[e] : 0, dv = ok ? dst[e] : 0;
    sb[wave][lo] = sv; db[wave][lo] = dv;
    zsb[wave][lo] = z[sv]; zdb[wave][lo] = z[dv];
  }
  {
    unsigned short* hrow = &hb[wave][lo * HS + hi * 32];
#pragma unroll
    for (int kk = 0; kk < 32; ++kk) {
      float mu = (float)(hi * 32 + kk) * (1.0f / 63.0f);
      float t = de - mu;
      hrow[kk] = f2bf(__expf(-3969.0f * t * t) * cutv);   // gamma = (F-1)^2
    }
  }
  wave_fence();

  // ---- GEMM1: h = sp(bf @ Wf1 + b1) ----
  v16bf a0 = load_a(&hb[wave][0], lo, hi, 0);
  v16bf a1 = load_a(&hb[wave][0], lo, hi, 32);
  wave_fence();
#pragma unroll
  for (int t = 0; t < 4; ++t) {
    v8f c = {};
    c = wmma_bf16(a0, load_b(w1, lo, hi, t, 0), c);
    c = wmma_bf16(a1, load_b(w1, lo, hi, t, 32), c);
    const int col = t * 16 + lo;
    const float bias = b1[col];
#pragma unroll
    for (int r8 = 0; r8 < 8; ++r8)                 // C/D: row = r8+hi*8, col = lo
      hb[wave][(r8 + hi * 8) * HS + col] = f2bf(softplus_f(c[r8] + bias));
  }
  wave_fence();

  // ---- GEMM2: filt = (h @ Wf2 + b2) * cut; message; atomic scatter ----
  v16bf h0 = load_a(&hb[wave][0], lo, hi, 0);
  v16bf h1 = load_a(&hb[wave][0], lo, hi, 32);
#pragma unroll
  for (int t = 0; t < 4; ++t) {
    v8f c = {};
    c = wmma_bf16(h0, load_b(w2, lo, hi, t, 0), c);
    c = wmma_bf16(h1, load_b(w2, lo, hi, t, 32), c);
    const int col = t * 16 + lo;
    const float bias = b2[col];
#pragma unroll
    for (int r8 = 0; r8 < 8; ++r8) {
      const int row = r8 + hi * 8;
      const int ee = e0 + row;
      if (ee < E) {
        float filt = (c[r8] + bias) * cutb[wave][row];
        float eh = Ecolor[zsb[wave][row] * 64 + col] + Ecolor[zdb[wave][row] * 64 + col];
        float val = filt * eh;
        if (y) val *= y[(size_t)sb[wave][row] * 64 + col];
        atomicAdd(&agg[(size_t)db[wave][row] * 64 + col], val);
      }
    }
  }
}

// ---------------------------------------------------------------------------
// Node MLP: out = [resid +] (W2t ? sp(in@W1+b1)@W2+b2 : in@W1(+b1))
// ---------------------------------------------------------------------------
__global__ __launch_bounds__(256) void node_mlp(
    const float* __restrict__ in, const unsigned short* __restrict__ W1t,
    const float* __restrict__ b1, const unsigned short* __restrict__ W2t,
    const float* __restrict__ b2, float* __restrict__ out,
    const float* __restrict__ resid, int N) {
  __shared__ unsigned short w1[4096];
  __shared__ unsigned short w2[4096];
  __shared__ unsigned short hb[WPB][16 * HS];

  for (int i = threadIdx.x; i < 4096; i += 256) { w1[i] = W1t[i]; if (W2t) w2[i] = W2t[i]; }
  __syncthreads();

  const int wave = threadIdx.x >> 5, lane = threadIdx.x & 31;
  const int lo = lane & 15, hi = lane >> 4;
  const int row0 = (blockIdx.x * WPB + wave) * 16;

  {   // stage A tile f32 -> bf16 LDS (row = lo, K half = hi)
    const int r = row0 + lo;
    const bool ok = (r < N);
    const float* srcp = in + (size_t)(ok ? r : 0) * 64 + hi * 32;
    unsigned short* dstp = &hb[wave][lo * HS + hi * 32];
#pragma unroll
    for (int kk = 0; kk < 32; ++kk) dstp[kk] = f2bf(ok ? srcp[kk] : 0.0f);
  }
  wave_fence();

  v16bf a0 = load_a(&hb[wave][0], lo, hi, 0);
  v16bf a1 = load_a(&hb[wave][0], lo, hi, 32);
  wave_fence();

  if (!W2t) {
#pragma unroll
    for (int t = 0; t < 4; ++t) {
      v8f c = {};
      c = wmma_bf16(a0, load_b(w1, lo, hi, t, 0), c);
      c = wmma_bf16(a1, load_b(w1, lo, hi, t, 32), c);
      const int col = t * 16 + lo;
      const float bias = b1 ? b1[col] : 0.0f;
#pragma unroll
      for (int r8 = 0; r8 < 8; ++r8) {
        const int row = row0 + r8 + hi * 8;
        if (row < N) {
          float v = c[r8] + bias;
          if (resid) v += resid[(size_t)row * 64 + col];
          out[(size_t)row * 64 + col] = v;
        }
      }
    }
  } else {
#pragma unroll
    for (int t = 0; t < 4; ++t) {
      v8f c = {};
      c = wmma_bf16(a0, load_b(w1, lo, hi, t, 0), c);
      c = wmma_bf16(a1, load_b(w1, lo, hi, t, 32), c);
      const int col = t * 16 + lo;
      const float bias = b1[col];
#pragma unroll
      for (int r8 = 0; r8 < 8; ++r8)
        hb[wave][(r8 + hi * 8) * HS + col] = f2bf(softplus_f(c[r8] + bias));
    }
    wave_fence();
    v16bf h0 = load_a(&hb[wave][0], lo, hi, 0);
    v16bf h1 = load_a(&hb[wave][0], lo, hi, 32);
#pragma unroll
    for (int t = 0; t < 4; ++t) {
      v8f c = {};
      c = wmma_bf16(h0, load_b(w2, lo, hi, t, 0), c);
      c = wmma_bf16(h1, load_b(w2, lo, hi, t, 32), c);
      const int col = t * 16 + lo;
      const float bias = b2[col];
#pragma unroll
      for (int r8 = 0; r8 < 8; ++r8) {
        const int row = row0 + r8 + hi * 8;
        if (row < N) {
          float v = c[r8] + bias;
          if (resid) v += resid[(size_t)row * 64 + col];
          out[(size_t)row * 64 + col] = v;
        }
      }
    }
  }
}

// ---------------------------------------------------------------------------
// Mean pooling + tiny classification head.
// ---------------------------------------------------------------------------
__global__ __launch_bounds__(256) void pool_sum(const float* __restrict__ xf,
                                                const int* __restrict__ gid,
                                                float* __restrict__ sums,
                                                float* __restrict__ cnt, int N) {
  int i = blockIdx.x * 256 + threadIdx.x;
  int node = i >> 6, col = i & 63;
  if (node < N) {
    int g = gid[node];
    atomicAdd(&sums[(size_t)g * 64 + col], xf[(size_t)node * 64 + col]);
    if (col == 0) atomicAdd(&cnt[g], 1.0f);
  }
}

__global__ __launch_bounds__(256) void pool_head(const float* __restrict__ sums,
                                                 const float* __restrict__ cnt,
                                                 const float* __restrict__ Wfc2,
                                                 const float* __restrict__ bfc2,
                                                 float* __restrict__ out, int G, int C) {
  int i = blockIdx.x * 256 + threadIdx.x;
  if (i >= G * C) return;
  int g = i / C, c = i % C;
  float m = cnt[g]; m = (m > 1.0f) ? m : 1.0f;
  float acc = bfc2[c];
#pragma unroll
  for (int k = 0; k < 64; ++k) acc += (sums[g * 64 + k] / m) * Wfc2[k * C + c];
  out[i] = acc;
}

// ---------------------------------------------------------------------------
extern "C" void kernel_launch(void* const* d_in, const int* in_sizes, int n_in,
                              void* d_out, int out_size, void* d_ws, size_t ws_size,
                              hipStream_t stream) {
  const float* d_dist  = (const float*)d_in[0];
  const int*   src     = (const int*)  d_in[1];
  const int*   dst     = (const int*)  d_in[2];
  const int*   z       = (const int*)  d_in[3];
  const int*   gid     = (const int*)  d_in[4];
  const float* Ecolor  = (const float*)d_in[5];
  const float* Wf1e = (const float*)d_in[6],  *b1e = (const float*)d_in[7];
  const float* Wf2e = (const float*)d_in[8],  *b2e = (const float*)d_in[9];
  const float* We   = (const float*)d_in[10], *be  = (const float*)d_in[11];
  const float* Wf1  = (const float*)d_in[12], *b1  = (const float*)d_in[13];
  const float* Wf2  = (const float*)d_in[14], *b2  = (const float*)d_in[15];
  const float* Win  = (const float*)d_in[16];
  const float* Wo1  = (const float*)d_in[17], *bo1 = (const float*)d_in[18];
  const float* Wo2  = (const float*)d_in[19], *bo2 = (const float*)d_in[20];
  const float* Wfc  = (const float*)d_in[21], *bfc = (const float*)d_in[22];
  const float* Wfc2 = (const float*)d_in[23], *bfc2= (const float*)d_in[24];

  const int E = in_sizes[0];
  const int N = in_sizes[3];
  const int L = in_sizes[12] / (64 * 64);
  const int C = in_sizes[24];
  const int G = out_size / C;

  // workspace layout
  char* ws = (char*)d_ws;
  const size_t n64 = (size_t)N * 64;
  unsigned short* wt = (unsigned short*)(ws);           // 19 * 4096 bf16
  size_t off = ((size_t)19 * 4096 * 2 + 255) & ~(size_t)255;
  float* ybuf = (float*)(ws + off);  off += ((n64 * 4 + 255) & ~(size_t)255);
  float* agg  = (float*)(ws + off);  off += ((n64 * 4 + 255) & ~(size_t)255);
  float* x    = (float*)(ws + off);  off += ((n64 * 4 + 255) & ~(size_t)255);
  float* sums = (float*)(ws + off);
  float* cnt  = (float*)(ws + off + (size_t)G * 64 * 4);
  const size_t poolBytes = (size_t)G * 64 * 4 + (size_t)G * 4;

  // weight transpose-convert (indices: 0 Wf1e, 1 Wf2e, 2 We, 3+l Wf1, 6+l Wf2,
  // 9+l Win, 12+l Wo1, 15+l Wo2, 18 Wfc)
  WSrc P;
  P.p[0] = Wf1e; P.p[1] = Wf2e; P.p[2] = We;
  for (int l = 0; l < 3; ++l) {
    P.p[3 + l]  = Wf1 + (size_t)l * 4096;
    P.p[6 + l]  = Wf2 + (size_t)l * 4096;
    P.p[9 + l]  = Win + (size_t)l * 4096;
    P.p[12 + l] = Wo1 + (size_t)l * 4096;
    P.p[15 + l] = Wo2 + (size_t)l * 4096;
  }
  P.p[18] = Wfc;
  hipLaunchKernelGGL(prep_weights, dim3(19), dim3(256), 0, stream, P, wt);

  const int etiles = (E + 15) / 16, eblocks = (etiles + WPB - 1) / WPB;
  const int ntiles = (N + 15) / 16, nblocks = (ntiles + WPB - 1) / WPB;

  // SchnetEmbedding: agg = segment_sum(eh * filt_e, dst); x = agg@We + be
  hipMemsetAsync(agg, 0, n64 * 4, stream);
  hipLaunchKernelGGL(edge_cfconv, dim3(eblocks), dim3(256), 0, stream,
                     d_dist, src, dst, z, Ecolor,
                     wt + 0 * 4096, b1e, wt + 1 * 4096, b2e,
                     (const float*)nullptr, agg, E);
  hipLaunchKernelGGL(node_mlp, dim3(nblocks), dim3(256), 0, stream,
                     agg, wt + 2 * 4096, be,
                     (const unsigned short*)nullptr, (const float*)nullptr,
                     x, (const float*)nullptr, N);

  // residual SchnetConv stack
  for (int l = 0; l < L; ++l) {
    hipLaunchKernelGGL(node_mlp, dim3(nblocks), dim3(256), 0, stream,
                       x, wt + (size_t)(9 + l) * 4096, (const float*)nullptr,
                       (const unsigned short*)nullptr, (const float*)nullptr,
                       ybuf, (const float*)nullptr, N);                  // y = x@Win
    hipMemsetAsync(agg, 0, n64 * 4, stream);
    hipLaunchKernelGGL(edge_cfconv, dim3(eblocks), dim3(256), 0, stream,
                       d_dist, src, dst, z, Ecolor,
                       wt + (size_t)(3 + l) * 4096, b1 + (size_t)l * 64,
                       wt + (size_t)(6 + l) * 4096, b2 + (size_t)l * 64,
                       ybuf, agg, E);
    hipLaunchKernelGGL(node_mlp, dim3(nblocks), dim3(256), 0, stream,
                       agg, wt + (size_t)(12 + l) * 4096, bo1 + (size_t)l * 64,
                       wt + (size_t)(15 + l) * 4096, bo2 + (size_t)l * 64,
                       x, x, N);                                         // x += MLP(agg)
  }

  // final node linear (reuse ybuf) + mean pooling + head
  hipLaunchKernelGGL(node_mlp, dim3(nblocks), dim3(256), 0, stream,
                     x, wt + (size_t)18 * 4096, bfc,
                     (const unsigned short*)nullptr, (const float*)nullptr,
                     ybuf, (const float*)nullptr, N);
  hipMemsetAsync(sums, 0, poolBytes, stream);
  hipLaunchKernelGGL(pool_sum, dim3((int)((n64 + 255) / 256)), dim3(256), 0, stream,
                     ybuf, gid, sums, cnt, N);
  hipLaunchKernelGGL(pool_head, dim3((G * C + 255) / 256), dim3(256), 0, stream,
                     sums, cnt, Wfc2, bfc2, (float*)d_out, G, C);
}
